// SimAttn_87144886436263
// MI455X (gfx1250) — matrix operand order repaired and checked by reading, not compiled, not run on hardware
//
#include <hip/hip_runtime.h>
#include <math.h>

// ---- types for WMMA fragments (gfx1250, wave32) ----
typedef __attribute__((ext_vector_type(16))) __bf16         v16bf;
typedef __attribute__((ext_vector_type(8)))  float          v8f;
typedef __attribute__((ext_vector_type(8)))  unsigned short v8us;

union Frag16 {
    struct { v8us lo; v8us hi; } u;
    v16bf bf;
};

__device__ __forceinline__ unsigned short f2bf(float f) {
    unsigned int u = __float_as_uint(f);
    u += 0x7FFFu + ((u >> 16) & 1u);   // round-to-nearest-even
    return (unsigned short)(u >> 16);
}

// ============================================================================
// Wave-level bf16 GEMM:  O[M,N] = A[M,K] @ BT[N,K]^T  (+bias over cols)
// One wave computes a 32x32 output macro-tile = 2x2 grid of 16x16 WMMA tiles
// (v_wmma_f32_16x16x32_bf16), so 8 b128 loads feed 4 WMMAs per k-step
// (2 loads/WMMA vs 4 for an unblocked tile).
// A row-major [M][K] bf16 (as u16); BT row-major [N][K] bf16.
// Fragment layouts per CDNA5 ISA 7.12.2:
//   A: lane m(0..15)/(16..31 = K-high half), 16 halves j -> k = j + 8h + (j>=8)*8
//   B: lane n holds column n, halves j -> k = j + 8h + (j>=8)*8 (same as A)
//   C/D: lane = n + 16*(m>=8); vgpr r -> M = r + 8h, N = lane&15
// Batched via gridDim.y with element strides sA/sBT/sO/sOb.
// M and N must be multiples of 32; K a multiple of 32.
// ============================================================================
template<bool HAS_BIAS, bool OUT_F32, bool OUT_BF, bool BF_TRANS>
__global__ __launch_bounds__(128)
void gemm_wmma_bf16(const unsigned short* __restrict__ A,
                    const unsigned short* __restrict__ BT,
                    const float* __restrict__ bias,
                    float* __restrict__ outF,
                    unsigned short* __restrict__ outB,
                    int M, int N, int K,
                    int lda, int ldbt, int ldo, int ldob,
                    long sA, long sBT, long sO, long sOb)
{
    const int lane = threadIdx.x & 31;
    const int wave = threadIdx.x >> 5;
    const int tilesN = N >> 5;                       // 32-wide macro-tiles
    const int tilesM = M >> 5;
    const int tile = blockIdx.x * 4 + wave;          // wave-uniform
    if (tile >= tilesM * tilesN) return;             // uniform exit: EXEC stays all-1
    const int tm = (tile / tilesN) * 32;
    const int tn = (tile % tilesN) * 32;

    A  += (long)blockIdx.y * sA;
    BT += (long)blockIdx.y * sBT;

    const int hm = lane >> 4;
    const int ln = lane & 15;

    const unsigned short* arow[2];
    const unsigned short* brow[2];
    arow[0] = A  + (size_t)(tm + ln) * lda;
    arow[1] = arow[0] + (size_t)16 * lda;
    brow[0] = BT + (size_t)(tn + ln) * ldbt;
    brow[1] = brow[0] + (size_t)16 * ldbt;

    v8f acc[2][2];
    acc[0][0] = (v8f){}; acc[0][1] = (v8f){};
    acc[1][0] = (v8f){}; acc[1][1] = (v8f){};

    for (int kb = 0; kb < K; kb += 32) {
        Frag16 af[2], bfr[2];
#pragma unroll
        for (int i = 0; i < 2; ++i) {
            af[i].u.lo  = *(const v8us*)(arow[i] + kb + 8 * hm);       // k=kb+8h..+7
            af[i].u.hi  = *(const v8us*)(arow[i] + kb + 16 + 8 * hm);  // k=kb+16+8h..+7
            bfr[i].u.lo = *(const v8us*)(brow[i] + kb + 8 * hm);       // k=kb+8h..+7
            bfr[i].u.hi = *(const v8us*)(brow[i] + kb + 16 + 8 * hm);  // k=kb+16+8h..+7
        }
#pragma unroll
        for (int i = 0; i < 2; ++i)
#pragma unroll
            for (int j = 0; j < 2; ++j)
                acc[i][j] = __builtin_amdgcn_wmma_f32_16x16x32_bf16(
                    false, af[i].bf, false, bfr[j].bf, (short)0, acc[i][j],
                    false, false);
    }

#pragma unroll
    for (int j = 0; j < 2; ++j) {
        const int col = tn + j * 16 + ln;
        const float bvs = HAS_BIAS ? bias[col] : 0.0f;
#pragma unroll
        for (int i = 0; i < 2; ++i) {
#pragma unroll
            for (int r = 0; r < 8; ++r) {
                const int row = tm + i * 16 + 8 * hm + r;
                const float v = acc[i][j][r] + bvs;
                if (OUT_F32)
                    outF[(long)blockIdx.y * sO + (size_t)row * ldo + col] = v;
                if (OUT_BF) {
                    if (BF_TRANS)
                        outB[(long)blockIdx.y * sOb + (size_t)col * ldob + row] = f2bf(v);
                    else
                        outB[(long)blockIdx.y * sOb + (size_t)row * ldob + col] = f2bf(v);
                }
            }
        }
    }
}

// ---- prep kernels -----------------------------------------------------------
__global__ void k_f32_to_bf16(const float* __restrict__ src,
                              unsigned short* __restrict__ dst, int n) {
    int i = blockIdx.x * blockDim.x + threadIdx.x;
    if (i < n) dst[i] = f2bf(src[i]);
}

// dst[N][K] = bf16(src[K][N])
__global__ void k_transpose_bf16(const float* __restrict__ src,
                                 unsigned short* __restrict__ dst, int K, int N) {
    int i = blockIdx.x * blockDim.x + threadIdx.x;
    if (i >= K * N) return;
    int k = i % K, n = i / K;
    dst[i] = f2bf(src[(size_t)k * N + n]);
}

// WqbT[n][k] = Wq[k][n] + sum_c Wl0[k][c*512+n]
__global__ void k_make_wqbT(const float* __restrict__ Wq,
                            const float* __restrict__ Wl0,
                            unsigned short* __restrict__ dst) {
    int i = blockIdx.x * blockDim.x + threadIdx.x;
    if (i >= 512 * 512) return;
    int k = i % 512, n = i / 512;
    float v = Wq[(size_t)k * 512 + n];
#pragma unroll
    for (int c = 0; c < 5; ++c) v += Wl0[(size_t)k * 2560 + c * 512 + n];
    dst[i] = f2bf(v);
}

// dst[m][k] = X[k][m] + s2*Y[k][m],  m in [0,2560), k in [0,512)
__global__ void k_combineT(const float* __restrict__ X,
                           const float* __restrict__ Y, float s2,
                           unsigned short* __restrict__ dst) {
    int i = blockIdx.x * blockDim.x + threadIdx.x;
    if (i >= 2560 * 512) return;
    int k = i % 512, m = i / 512;
    dst[i] = f2bf(X[(size_t)k * 2560 + m] + s2 * Y[(size_t)k * 2560 + m]);
}

// QX[b][i][t][k]: i==0 -> base[(b*192+t)][k]; else diff[(b*192+t)][(i-1)*512+k]
__global__ void k_pack6(const unsigned short* __restrict__ base,
                        const unsigned short* __restrict__ diff,
                        unsigned short* __restrict__ QX) {
    int idx = blockIdx.x * blockDim.x + threadIdx.x;
    const int total = 4 * 6 * 192 * 512;
    if (idx >= total) return;
    int k = idx & 511;
    int t = (idx >> 9) % 192;
    int i = (idx / (512 * 192)) % 6;
    int b = idx / (512 * 192 * 6);
    QX[idx] = (i == 0) ? base[((size_t)(b * 192 + t)) * 512 + k]
                       : diff[((size_t)(b * 192 + t)) * 2560 + (size_t)(i - 1) * 512 + k];
}

// ---- score combine (u^T G u') + masked softmax -> P (bf16) ------------------
__global__ __launch_bounds__(256)
void k_score_softmax(const float* __restrict__ G, const float* __restrict__ qa,
                     const float* __restrict__ aq, const int* __restrict__ mask,
                     unsigned short* __restrict__ Pbf) {
    const int bt = blockIdx.x;                 // 0..767 == (b,t)
    const int b = bt / 192, t = bt % 192;
    const int s = threadIdx.x;
    __shared__ float red[256];
    const bool valid = s < 192;
    float sc = -1e30f;
    if (valid) {
        float uq[5], ua[5];
#pragma unroll
        for (int c = 0; c < 5; ++c) {
            uq[c] = qa[(((size_t)b * 192 + t) * 192 + s) * 5 + c];
            ua[c] = aq[(((size_t)b * 192 + s) * 192 + t) * 5 + c];
        }
        float total = 0.0f;
#pragma unroll
        for (int i = 0; i < 6; ++i) {
            const float* grow = G + ((size_t)b * 1152 + i * 192 + t) * 1152 + s;
            float acc = grow[0];
#pragma unroll
            for (int j = 1; j < 6; ++j) acc += ua[j - 1] * grow[(size_t)j * 192];
            total += (i == 0) ? acc : uq[i - 1] * acc;
        }
        sc = total * 0.04419417382415922f;     // 1/sqrt(512)
        if (mask[b * 192 + s] == 0) sc = -1e9f;
    }
    red[threadIdx.x] = sc;
    __syncthreads();
    for (int off = 128; off > 0; off >>= 1) {
        if (threadIdx.x < off)
            red[threadIdx.x] = fmaxf(red[threadIdx.x], red[threadIdx.x + off]);
        __syncthreads();
    }
    const float m = red[0];
    __syncthreads();
    const float p = valid ? __expf(sc - m) : 0.0f;
    red[threadIdx.x] = p;
    __syncthreads();
    for (int off = 128; off > 0; off >>= 1) {
        if (threadIdx.x < off) red[threadIdx.x] += red[threadIdx.x + off];
        __syncthreads();
    }
    if (valid) Pbf[(size_t)bt * 192 + s] = f2bf(p / red[0]);
}

// ---- LayerNorm over last dim (512) -----------------------------------------
__global__ __launch_bounds__(256)
void k_layernorm(const float* __restrict__ X, const float* __restrict__ w,
                 const float* __restrict__ bb, float* __restrict__ out) {
    const int row = blockIdx.x;
    const float* x = X + (size_t)row * 512;
    __shared__ float s1[256], s2[256];
    const float a = x[threadIdx.x], c = x[threadIdx.x + 256];
    s1[threadIdx.x] = a + c;
    s2[threadIdx.x] = a * a + c * c;
    __syncthreads();
    for (int off = 128; off > 0; off >>= 1) {
        if (threadIdx.x < off) {
            s1[threadIdx.x] += s1[threadIdx.x + off];
            s2[threadIdx.x] += s2[threadIdx.x + off];
        }
        __syncthreads();
    }
    const float mean = s1[0] * (1.0f / 512.0f);
    const float var  = s2[0] * (1.0f / 512.0f) - mean * mean;
    const float inv  = rsqrtf(var + 1e-12f);
    float* o = out + (size_t)row * 512;
    o[threadIdx.x]       = w[threadIdx.x]       * ((a - mean) * inv) + bb[threadIdx.x];
    o[threadIdx.x + 256] = w[threadIdx.x + 256] * ((c - mean) * inv) + bb[threadIdx.x + 256];
}

// ============================================================================
extern "C" void kernel_launch(void* const* d_in, const int* in_sizes, int n_in,
                              void* d_out, int out_size, void* d_ws, size_t ws_size,
                              hipStream_t stream) {
    const float* query = (const float*)d_in[0];
    const float* key_  = (const float*)d_in[1];
    const float* value = (const float*)d_in[2];
    const float* qa    = (const float*)d_in[3];
    const float* aq    = (const float*)d_in[4];
    const int*   mask  = (const int*)d_in[5];
    const float* Wq  = (const float*)d_in[6];  const float* bq = (const float*)d_in[7];
    const float* Wk  = (const float*)d_in[8];  const float* bk = (const float*)d_in[9];
    const float* Wv  = (const float*)d_in[10]; const float* bv = (const float*)d_in[11];
    const float* Wl1 = (const float*)d_in[12];
    const float* Wr1 = (const float*)d_in[13];
    const float* Wl0 = (const float*)d_in[14];
    const float* Wr0 = (const float*)d_in[15];
    const float* Wo  = (const float*)d_in[16]; const float* bo = (const float*)d_in[17];
    const float* lnw = (const float*)d_in[18];
    const float* lnb = (const float*)d_in[19];
    float* out = (float*)d_out;

    // ---- workspace carve-out (~53 MB) ----
    char* ws = (char*)d_ws;
    size_t off = 0;
    auto aus = [&](size_t n) { unsigned short* p = (unsigned short*)(ws + off);
                               off += ((n * 2 + 255) / 256) * 256; return p; };
    auto af  = [&](size_t n) { float* p = (float*)(ws + off);
                               off += ((n * 4 + 255) / 256) * 256; return p; };

    unsigned short* qbf    = aus(768 * 512);
    unsigned short* kbf    = aus(768 * 512);
    unsigned short* vbf    = aus(768 * 512);
    unsigned short* WqbT   = aus(512 * 512);
    unsigned short* WkT    = aus(512 * 512);
    unsigned short* WvT    = aus(512 * 512);
    unsigned short* WoT    = aus(512 * 512);
    unsigned short* WqdT   = aus((size_t)2560 * 512);
    unsigned short* WksT   = aus((size_t)2560 * 512);
    unsigned short* qbase  = aus(768 * 512);
    unsigned short* qdiff  = aus((size_t)768 * 2560);
    unsigned short* klin   = aus(768 * 512);
    unsigned short* ksum   = aus((size_t)768 * 2560);
    unsigned short* vprojT = aus((size_t)4 * 512 * 192);
    unsigned short* QX     = aus((size_t)4 * 6 * 192 * 512);
    unsigned short* KX     = aus((size_t)4 * 6 * 192 * 512);
    float*          G      = af((size_t)4 * 1152 * 1152);
    unsigned short* Pbf    = aus((size_t)4 * 192 * 192);
    unsigned short* ctxbf  = aus(768 * 512);
    float*          lnin   = af(768 * 512);

    const int TPB = 256;
    auto g1 = [&](int n) { return dim3((n + TPB - 1) / TPB); };
    auto gg = [&](int M, int N, int batches) {
        int tiles = (M / 32) * (N / 32);          // 32x32 macro-tiles, 1 per wave
        return dim3((tiles + 3) / 4, batches);
    };

    // activations -> bf16
    k_f32_to_bf16<<<g1(768 * 512), TPB, 0, stream>>>(query, qbf, 768 * 512);
    k_f32_to_bf16<<<g1(768 * 512), TPB, 0, stream>>>(key_,  kbf, 768 * 512);
    k_f32_to_bf16<<<g1(768 * 512), TPB, 0, stream>>>(value, vbf, 768 * 512);
    // weight transposes/combines -> bf16 [N][K]
    k_transpose_bf16<<<g1(512 * 512), TPB, 0, stream>>>(Wk, WkT, 512, 512);
    k_transpose_bf16<<<g1(512 * 512), TPB, 0, stream>>>(Wv, WvT, 512, 512);
    k_transpose_bf16<<<g1(512 * 512), TPB, 0, stream>>>(Wo, WoT, 512, 512);
    k_make_wqbT<<<g1(512 * 512), TPB, 0, stream>>>(Wq, Wl0, WqbT);
    k_combineT<<<g1(2560 * 512), TPB, 0, stream>>>(Wl1, Wl0, -1.0f, WqdT);  // qdiff weights
    k_combineT<<<g1(2560 * 512), TPB, 0, stream>>>(Wr0, Wr1,  1.0f, WksT);  // ksum  weights

    // projections (WMMA): qbase, qdiff, klin, ksum, vprojT
    gemm_wmma_bf16<true,  false, true, false><<<gg(768, 512, 1), 128, 0, stream>>>(
        qbf, WqbT, bq, nullptr, qbase, 768, 512, 512, 512, 512, 0, 512, 0, 0, 0, 0);
    gemm_wmma_bf16<false, false, true, false><<<gg(768, 2560, 1), 128, 0, stream>>>(
        qbf, WqdT, nullptr, nullptr, qdiff, 768, 2560, 512, 512, 512, 0, 2560, 0, 0, 0, 0);
    gemm_wmma_bf16<true,  false, true, false><<<gg(768, 512, 1), 128, 0, stream>>>(
        kbf, WkT, bk, nullptr, klin, 768, 512, 512, 512, 512, 0, 512, 0, 0, 0, 0);
    gemm_wmma_bf16<false, false, true, false><<<gg(768, 2560, 1), 128, 0, stream>>>(
        kbf, WksT, nullptr, nullptr, ksum, 768, 2560, 512, 512, 512, 0, 2560, 0, 0, 0, 0);
    gemm_wmma_bf16<true,  false, true, true ><<<gg(192, 512, 4), 128, 0, stream>>>(
        vbf, WvT, bv, nullptr, vprojT, 192, 512, 512, 512, 512, 0, 192,
        (long)192 * 512, 0, 0, (long)512 * 192);

    // stack [qbase|qdiff] and [klin|ksum] into [B][6*192][512]
    k_pack6<<<g1(4 * 6 * 192 * 512), TPB, 0, stream>>>(qbase, qdiff, QX);
    k_pack6<<<g1(4 * 6 * 192 * 512), TPB, 0, stream>>>(klin, ksum, KX);

    // G[b] = QX[b] @ KX[b]^T  (batched WMMA, f32 out)
    gemm_wmma_bf16<false, true, false, false><<<gg(1152, 1152, 4), 128, 0, stream>>>(
        QX, KX, nullptr, G, nullptr, 1152, 1152, 512, 512, 512, 1152, 0,
        (long)1152 * 512, (long)1152 * 512, (long)1152 * 1152, 0);

    // scores = u^T G u' / sqrt(D), mask, softmax -> P (bf16)
    k_score_softmax<<<dim3(768), 256, 0, stream>>>(G, qa, aq, mask, Pbf);

    // ctx[b] = P[b] @ vproj[b]  (batched WMMA, bf16 out)
    gemm_wmma_bf16<false, false, true, false><<<gg(192, 512, 4), 128, 0, stream>>>(
        Pbf, vprojT, nullptr, nullptr, ctxbf, 192, 512, 192, 192, 192, 0, 512,
        (long)192 * 192, (long)512 * 192, 0, (long)192 * 512);

    // out = ctx @ Wo + bo (f32), then LayerNorm
    gemm_wmma_bf16<true, true, false, false><<<gg(768, 512, 1), 128, 0, stream>>>(
        ctxbf, WoT, bo, lnin, nullptr, 768, 512, 512, 512, 512, 512, 0, 0, 0, 0, 0);
    k_layernorm<<<dim3(768), 256, 0, stream>>>(lnin, lnw, lnb, out);

    (void)in_sizes; (void)n_in; (void)out_size; (void)ws_size;
}
